// DTS_SNN_2D_18648747999261
// MI455X (gfx1250) — matrix operand order repaired and checked by reading, not compiled, not run on hardware
//
#include <hip/hip_runtime.h>

typedef __attribute__((ext_vector_type(16))) _Float16 v16h;
typedef __attribute__((ext_vector_type(8)))  _Float16 v8h;
typedef __attribute__((ext_vector_type(8)))  float    v8f;
typedef __attribute__((ext_vector_type(4)))  float    f32x4;
typedef __attribute__((ext_vector_type(4)))  int      i32x4;

// Problem constants (match reference)
constexpr int Hh = 128, Ww = 128, Bb = 16, Tt = 60;
constexpr int HID = 512, OUTN = 11, INPUTS = 9216;   // 9*32*32
constexpr int KSTEPS = INPUTS / 32;                  // 288 k-steps of 32
constexpr float D1 = 0.9512294245007140f;            // exp(-1/20)
constexpr float D2 = 0.8187307530779818f;            // exp(-1/5)
constexpr float DMc = 0.9512294245007140f;           // exp(-1/20)
constexpr float DSc = 0.8187307530779818f;           // exp(-1/5)
constexpr float THRESH = 0.3f;
constexpr float SCALEc = 0.5f;

// Launch shape: persistent kernel, 32 workgroups x 256 threads (8 wave32s)
#define NBLK   32
#define NTHR   256
#define NWAVE  8
#define TOTTHR (NBLK * NTHR)
constexpr int KS_PER_WAVE = KSTEPS / NWAVE;          // 36

// ---------------------------------------------------------------------------
// Weight packing: w_hid (HID x INPUTS, f32) -> f16 in this file's contiguous-k B-operand
// layout. One 16x16x32 B tile = 32 lanes x 16 halves (lane-major, v16h per
// lane): lanes 0-15 hold K=0..15 (vgpr v -> K=2v,2v+1), lanes 16-31 K=16..31.
// Block id = nt*KSTEPS + ks ; block stride = 512 halves (1 KB).
// ---------------------------------------------------------------------------
__global__ void pack_whid_kernel(const float* __restrict__ w_hid,
                                 _Float16* __restrict__ outp) {
  int flat = blockIdx.x * blockDim.x + threadIdx.x;
  if (flat >= (HID / 16) * KSTEPS * 512) return;
  int blockid = flat >> 9;
  int e = flat & 511;
  int lane = e >> 4;
  int idx16 = e & 15;
  int v = idx16 >> 1, hsel = idx16 & 1;
  int nt = blockid / KSTEPS, ks = blockid % KSTEPS;
  int n = nt * 16 + (lane & 15);
  int k = ks * 32 + ((lane >> 4) << 4) + v * 2 + hsel;
  outp[flat] = (_Float16)w_hid[n * INPUTS + k];
}

__global__ void pack_wout_kernel(const float* __restrict__ w_out,
                                 _Float16* __restrict__ outp) {
  int flat = blockIdx.x * blockDim.x + threadIdx.x;  // 16 ks-blocks * 512
  if (flat >= (HID / 32) * 512) return;
  int ks = flat >> 9;
  int e = flat & 511;
  int lane = e >> 4;
  int idx16 = e & 15;
  int v = idx16 >> 1, hsel = idx16 & 1;
  int n = lane & 15;
  int k = ks * 32 + ((lane >> 4) << 4) + v * 2 + hsel;
  outp[flat] = (n < OUTN) ? (_Float16)w_out[n * HID + k] : (_Float16)0.0f;
}

// ---------------------------------------------------------------------------
// Grid-wide barrier (persistent kernel; counter zeroed by memsetAsync per call)
// ---------------------------------------------------------------------------
__device__ __forceinline__ void grid_barrier(unsigned* ctr, unsigned* epoch_s) {
  __syncthreads();
  if (threadIdx.x == 0) {
    unsigned e = ++(*epoch_s);
    __threadfence();
    __hip_atomic_fetch_add(ctr, 1u, __ATOMIC_RELEASE, __HIP_MEMORY_SCOPE_AGENT);
    while (__hip_atomic_load(ctr, __ATOMIC_ACQUIRE, __HIP_MEMORY_SCOPE_AGENT) <
           e * (unsigned)NBLK) {
      __builtin_amdgcn_s_sleep(2);
    }
    __threadfence();
  }
  __syncthreads();
}

// ---------------------------------------------------------------------------
// Persistent SNN kernel: the whole 60-step scan in one launch.
// Phases per step: P1 trace update -> P2 conv+input LIF -> P3 hidden GEMM
// (WMMA) + hidden LIF -> P4 output GEMM (WMMA, block0/wave0) + output LIF.
// ---------------------------------------------------------------------------
__global__ void __launch_bounds__(NTHR) snn_persistent(
    const int* __restrict__ events, const float* __restrict__ w_conv,
    float* __restrict__ out, unsigned* ctr,
    float* tr1, float* tr2, float* tdiff, float* m_in,
    float* c_h, float* m_h,
    float* c_o, float* m_o, float* s_o, float* accb,
    _Float16* s_in16, _Float16* s_h16,
    const _Float16* __restrict__ whidp, const _Float16* __restrict__ woutp) {
  __shared__ float wconv_s[32];
  __shared__ float red[NWAVE * 256];
  __shared__ unsigned epoch_s;

  const int tid = threadIdx.x;
  const int gtid = blockIdx.x * NTHR + tid;
  const int lane = tid & 31;
  const int wave = tid >> 5;

  if (tid < 32) wconv_s[tid] = w_conv[tid];
  if (tid == 0) epoch_s = 0u;
  __syncthreads();

  for (int t = 0; t < Tt; ++t) {
    // ---------------- Phase 1: dual-trace update + tdiff (4-wide) ----------
    // Groups of 4 elements never cross a (b,c) plane (16384 % 4 == 0).
    for (int it = 0; it < (Bb * 2 * Hh * Ww) / (4 * TOTTHR); ++it) {
      int f4 = gtid + it * TOTTHR;           // quad index
      int f = f4 * 4;
      int b = f >> 15;                       // / (2*128*128)
      int rem = f & 32767;                   // c*16384 + h*128 + w
      i32x4 ev = *(const i32x4*)(events + (b * Tt + t) * 32768 + rem);
      f32x4 a1 = *(const f32x4*)(tr1 + f);
      f32x4 a2 = *(const f32x4*)(tr2 + f);
      f32x4 td;
      for (int z = 0; z < 4; ++z) {
        int e = ev[z]; e = e < 0 ? 0 : (e > 1 ? 1 : e);
        float ef = (float)e;
        a1[z] = a1[z] * D1 + ef;
        a2[z] = a2[z] * D2 + ef;
        td[z] = (a1[z] - a2[z]) * SCALEc;
      }
      *(f32x4*)(tr1 + f) = a1;
      *(f32x4*)(tr2 + f) = a2;
      *(f32x4*)(tdiff + f) = td;
    }
    grid_barrier(ctr, &epoch_s);

    // ---------------- Phase 2: blocked conv + input LIF --------------------
    for (int it = 0; it < (Bb * INPUTS) / TOTTHR; ++it) {
      int nidx = gtid + it * TOTTHR;
      int b = nidx / INPUTS;
      int k = nidx - b * INPUTS;
      int d = k >> 10;
      int ij = k & 1023;
      int i = ij >> 5, j = ij & 31;
      int di = d / 3 - 1, dj = d % 3 - 1;
      int r0 = i * 4 + di, c0 = j * 4 + dj;
      float enc = 0.f;
      for (int c = 0; c < 2; ++c) {
        const float* tp = tdiff + ((b * 2 + c) << 14);
        for (int p = 0; p < 4; ++p) {
          int rr = r0 + p;
          if (rr < 0 || rr >= Hh) continue;
          const float* rowp = tp + rr * Ww;
          for (int q = 0; q < 4; ++q) {
            int cc = c0 + q;
            if (cc < 0 || cc >= Ww) continue;
            enc += wconv_s[c * 16 + p * 4 + q] * rowp[cc];
          }
        }
      }
      float sprev = (float)s_in16[nidx];
      float m = m_in[nidx] * DMc * (1.f - sprev) + enc;
      m_in[nidx] = m;
      s_in16[nidx] = (_Float16)((m - THRESH) >= 0.f ? 1.f : 0.f);
    }
    grid_barrier(ctr, &epoch_s);

    // ---------------- Phase 3: hidden GEMM (WMMA) + hidden LIF -------------
    {
      const int nt = blockIdx.x;         // N-tile (16 hidden cols per WG)
      const int mrow = lane & 15;
      const int lo = (lane >> 4) << 4;   // 0 or 16 (A-layout K split)
      v8f acc = {};
      for (int ksi = 0; ksi < KS_PER_WAVE; ++ksi) {
        int ks = wave * KS_PER_WAVE + ksi;
        const _Float16* ap = s_in16 + mrow * INPUTS + ks * 32 + lo;
        v8h a0 = *(const v8h*)(ap);
        v8h a1 = *(const v8h*)(ap + 8);
        v16h amat;
        for (int z = 0; z < 8; ++z) { amat[z] = a0[z]; amat[8 + z] = a1[z]; }
        const _Float16* bp = whidp + (nt * KSTEPS + ks) * 512 + lane * 16;
        // WGP-scope prefetch of next B tile (locality 3 -> near cache)
        if (ksi + 1 < KS_PER_WAVE) __builtin_prefetch(bp + 512, 0, 3);
        v16h bmat = *(const v16h*)(bp);
        acc = __builtin_amdgcn_wmma_f32_16x16x32_f16(
            false, amat, false, bmat, (short)0, acc, false, false);
      }
      // cross-wave reduction of partial 16x16 tiles via LDS
      {
        int n = lane & 15;
        int hi8 = (lane >> 4) << 3;
        for (int v = 0; v < 8; ++v)
          red[wave * 256 + (v + hi8) * 16 + n] = acc[v];
      }
      __syncthreads();
      {
        float s = 0.f;
        for (int wv = 0; wv < NWAVE; ++wv) s += red[wv * 256 + tid];
        int m = tid >> 4, nn = tid & 15;
        int idx = m * HID + nt * 16 + nn;
        float ch = c_h[idx] * DSc + s;
        c_h[idx] = ch;
        float shp = (float)s_h16[idx];
        float mh = m_h[idx] * DMc * (1.f - shp) + ch;
        m_h[idx] = mh;
        s_h16[idx] = (_Float16)((mh - THRESH) >= 0.f ? 1.f : 0.f);
      }
      __syncthreads();
    }
    grid_barrier(ctr, &epoch_s);

    // ---------------- Phase 4: output GEMM (WMMA) + output LIF -------------
    // Only block 0 / wave 0. No trailing barrier needed: the next write to
    // s_h16 (phase 3 of t+1) sits behind two grid barriers that block 0 can
    // only reach after finishing this phase.
    if (blockIdx.x == 0 && wave == 0) {
      const int mrow = lane & 15;
      const int lo = (lane >> 4) << 4;
      v8f acc = {};
      for (int ks = 0; ks < HID / 32; ++ks) {
        const _Float16* ap = s_h16 + mrow * HID + ks * 32 + lo;
        v8h a0 = *(const v8h*)(ap);
        v8h a1 = *(const v8h*)(ap + 8);
        v16h amat;
        for (int z = 0; z < 8; ++z) { amat[z] = a0[z]; amat[8 + z] = a1[z]; }
        v16h bmat = *(const v16h*)(woutp + ks * 512 + lane * 16);
        acc = __builtin_amdgcn_wmma_f32_16x16x32_f16(
            false, amat, false, bmat, (short)0, acc, false, false);
      }
      int n = lane & 15;
      int hi8 = (lane >> 4) << 3;
      for (int v = 0; v < 8; ++v) {
        int m = v + hi8;
        int idx = m * 16 + n;                  // padded 16x16 state
        float co = c_o[idx] * DSc + acc[v];
        c_o[idx] = co;
        float sop = s_o[idx];
        float mo = m_o[idx] * DMc * (1.f - sop) + co;
        m_o[idx] = mo;
        float so = (mo - THRESH) >= 0.f ? 1.f : 0.f;
        s_o[idx] = so;
        float av = accb[idx] + so;
        accb[idx] = av;
        if (t == Tt - 1 && n < OUTN) out[m * OUTN + n] = av * (1.0f / Tt);
      }
    }
  }
}

// ---------------------------------------------------------------------------
extern "C" void kernel_launch(void* const* d_in, const int* in_sizes, int n_in,
                              void* d_out, int out_size, void* d_ws,
                              size_t ws_size, hipStream_t stream) {
  const int*   events = (const int*)d_in[0];
  const float* w_conv = (const float*)d_in[1];
  const float* w_hid  = (const float*)d_in[2];
  const float* w_out  = (const float*)d_in[3];
  (void)in_sizes; (void)n_in; (void)out_size; (void)ws_size;

  // Workspace carve-up (256 B aligned). Zero-init region first, then weights.
  char* ws = (char*)d_ws;
  size_t off = 0;
  auto carve = [&](size_t bytes) -> char* {
    char* p = ws + off;
    off = (off + bytes + 255) & ~(size_t)255;
    return p;
  };
  unsigned* ctr    = (unsigned*)carve(256);
  float* tr1       = (float*)carve(sizeof(float) * Bb * 2 * Hh * Ww);
  float* tr2       = (float*)carve(sizeof(float) * Bb * 2 * Hh * Ww);
  float* tdiff     = (float*)carve(sizeof(float) * Bb * 2 * Hh * Ww);
  float* m_in      = (float*)carve(sizeof(float) * Bb * INPUTS);
  float* c_h       = (float*)carve(sizeof(float) * Bb * HID);
  float* m_h       = (float*)carve(sizeof(float) * Bb * HID);
  float* c_o       = (float*)carve(sizeof(float) * Bb * 16);
  float* m_o       = (float*)carve(sizeof(float) * Bb * 16);
  float* s_o       = (float*)carve(sizeof(float) * Bb * 16);
  float* accb      = (float*)carve(sizeof(float) * Bb * 16);
  _Float16* s_in16 = (_Float16*)carve(sizeof(_Float16) * Bb * INPUTS);
  _Float16* s_h16  = (_Float16*)carve(sizeof(_Float16) * Bb * HID);
  size_t zero_bytes = off;                         // everything above: zeroed
  _Float16* whidp  = (_Float16*)carve(sizeof(_Float16) * (HID / 16) * KSTEPS * 512);
  _Float16* woutp  = (_Float16*)carve(sizeof(_Float16) * (HID / 32) * 512);

  hipMemsetAsync(d_ws, 0, zero_bytes, stream);

  {
    int total = (HID / 16) * KSTEPS * 512;
    pack_whid_kernel<<<(total + 255) / 256, 256, 0, stream>>>(w_hid, whidp);
  }
  {
    int total = (HID / 32) * 512;
    pack_wout_kernel<<<(total + 255) / 256, 256, 0, stream>>>(w_out, woutp);
  }

  snn_persistent<<<NBLK, NTHR, 0, stream>>>(
      events, w_conv, (float*)d_out, ctr, tr1, tr2, tdiff, m_in, c_h, m_h,
      c_o, m_o, s_o, accb, s_in16, s_h16, whidp, woutp);
}
